// ECN_35459249996330
// MI455X (gfx1250) — matrix operand order repaired and checked by reading, not compiled, not run on hardware
//
#include <hip/hip_runtime.h>
#include <hip/hip_fp16.h>
#include <stdint.h>

#define N_NODES  10000
#define N_EDGES  160000
#define N_GRAPHS 128
#define HID      25
#define BN_EPS   1e-5f

typedef __attribute__((ext_vector_type(16))) _Float16 v16h;
typedef __attribute__((ext_vector_type(8)))  float    v8f;

// ---------------------------------------------------------------------------
// degree (scatter count of dst)
__global__ void k_deg(const int* __restrict__ ei, float* __restrict__ deg) {
    int e = blockIdx.x * blockDim.x + threadIdx.x;
    if (e < N_EDGES) atomicAdd(&deg[ei[N_EDGES + e]], 1.0f);
}

// cast/pad node features to f16 [N,32]
__global__ void k_cast_x16(const float* __restrict__ x, _Float16* __restrict__ x16, int din) {
    int t = blockIdx.x * blockDim.x + threadIdx.x;
    if (t >= N_NODES * 32) return;
    int n = t >> 5, i = t & 31;
    x16[t] = (i < din) ? (_Float16)x[n * din + i] : (_Float16)0.0f;
}

// edge MLP stage 1: per-column sum/sumsq over E for batch-norm stats
__global__ void k_edge_stats(const float* __restrict__ ea, const float* __restrict__ w1,
                             const float* __restrict__ b1, float* __restrict__ stats) {
    __shared__ float s[2 * HID];
    if (threadIdx.x < 2 * HID) s[threadIdx.x] = 0.0f;
    __syncthreads();
    float ls[HID], lq[HID];
#pragma unroll
    for (int k = 0; k < HID; ++k) { ls[k] = 0.0f; lq[k] = 0.0f; }
    for (int e = blockIdx.x * blockDim.x + threadIdx.x; e < N_EDGES;
         e += gridDim.x * blockDim.x) {
        float a0 = ea[e * 3], a1 = ea[e * 3 + 1], a2 = ea[e * 3 + 2];
#pragma unroll
        for (int k = 0; k < HID; ++k) {
            float h = a0 * w1[k] + a1 * w1[HID + k] + a2 * w1[2 * HID + k] + b1[k];
            ls[k] += h; lq[k] += h * h;
        }
    }
#pragma unroll
    for (int k = 0; k < HID; ++k) {
        atomicAdd(&s[k], ls[k]);
        atomicAdd(&s[HID + k], lq[k]);
    }
    __syncthreads();
    if (threadIdx.x < 2 * HID) atomicAdd(&stats[threadIdx.x], s[threadIdx.x]);
}

// fold bn into scale/shift
__global__ void k_bn_final(const float* __restrict__ stats, const float* __restrict__ g,
                           const float* __restrict__ be, float* __restrict__ bnp) {
    int k = threadIdx.x;
    if (k < HID) {
        float mu  = stats[k] / (float)N_EDGES;
        float var = stats[HID + k] / (float)N_EDGES - mu * mu;
        float sc  = g[k] * rsqrtf(var + BN_EPS);
        bnp[k]      = sc;
        bnp[32 + k] = be[k] - mu * sc;
    }
}

// edge MLP stage 2: h16[E,32] = relu(bn(ea@w1+b1)), slot 25 = 1.0 (bias fold), 26..31 = 0
__global__ void k_edge_h16(const float* __restrict__ ea, const float* __restrict__ w1,
                           const float* __restrict__ b1, const float* __restrict__ bnp,
                           _Float16* __restrict__ h16) {
    int e = blockIdx.x * blockDim.x + threadIdx.x;
    if (e >= N_EDGES) return;
    float a0 = ea[e * 3], a1 = ea[e * 3 + 1], a2 = ea[e * 3 + 2];
    _Float16* out = h16 + (size_t)e * 32;
#pragma unroll
    for (int k = 0; k < HID; ++k) {
        float h = a0 * w1[k] + a1 * w1[HID + k] + a2 * w1[2 * HID + k] + b1[k];
        h = h * bnp[k] + bnp[32 + k];
        h = h > 0.0f ? h : 0.0f;
        out[k] = (_Float16)h;
    }
    out[25] = (_Float16)1.0f;
#pragma unroll
    for (int k = 26; k < 32; ++k) out[k] = (_Float16)0.0f;
}

// B element for the Z GEMM: B[i, col] with col = kcol*32 + o
__device__ inline float bval(int i, int kcol, int o, int din,
                             const float* __restrict__ w2, const float* __restrict__ b2) {
    if (i >= din) return 0.0f;
    if (kcol < HID)  return w2[(size_t)kcol * din * 32 + i * 32 + o];
    if (kcol == HID) return b2[i * 32 + o];
    return 0.0f;
}

// pack B [32 x 1024] f16 into WMMA-B lanes, contiguous k as in A: per (chunk c, lane L) 8 dwords,
// lane holds column n = c*16 + L%16, halves of K split by L/16
__global__ void k_pack_b(const float* __restrict__ w2, const float* __restrict__ b2,
                         uint32_t* __restrict__ bpk, int din) {
    int t = blockIdx.x * blockDim.x + threadIdx.x;
    if (t >= 64 * 32) return;
    int L = t & 31, c = t >> 5;
    int nl = L & 15, half = L >> 4;
    int col  = c * 16 + nl;
    int kcol = col >> 5, o = col & 31;
    uint32_t* dst = bpk + (size_t)t * 8;
#pragma unroll
    for (int v = 0; v < 8; ++v) {
        int i0 = half * 16 + 2 * v;
        float f0 = bval(i0,     kcol, o, din, w2, b2);
        float f1 = bval(i0 + 1, kcol, o, din, w2, b2);
        unsigned short u0 = __builtin_bit_cast(unsigned short, (_Float16)f0);
        unsigned short u1 = __builtin_bit_cast(unsigned short, (_Float16)f1);
        dst[v] = ((uint32_t)u1 << 16) | (uint32_t)u0;
    }
}

// Z[N,1024] = X16[N,32] @ B[32,1024] via v_wmma_f32_16x16x32_f16
// block = 8 waves; block b = node tile b (16 nodes); wave w handles col chunks w*8..w*8+7
__global__ void __launch_bounds__(256) k_z_wmma(const _Float16* __restrict__ x16,
                                                const uint32_t* __restrict__ bpk,
                                                _Float16* __restrict__ z) {
    int tile = blockIdx.x;
    int wave = threadIdx.x >> 5;
    int lane = threadIdx.x & 31;
    int m    = lane & 15;     // A row within tile / D column within chunk
    int half = lane >> 4;

    // A: lanes 0-15 hold K 0-15, lanes 16-31 hold K 16-31 (pairs per VGPR)
    const uint4* row = reinterpret_cast<const uint4*>(x16 + ((size_t)tile * 16 + m) * 32);
    uint4 t0 = row[2 * half];
    uint4 t1 = row[2 * half + 1];
    union { v16h h; uint32_t u[8]; } A;
    A.u[0] = t0.x; A.u[1] = t0.y; A.u[2] = t0.z; A.u[3] = t0.w;
    A.u[4] = t1.x; A.u[5] = t1.y; A.u[6] = t1.z; A.u[7] = t1.w;

#pragma unroll
    for (int q = 0; q < 8; ++q) {
        int c = wave * 8 + q;
        const uint4* bp = reinterpret_cast<const uint4*>(bpk + ((size_t)c * 32 + lane) * 8);
        uint4 b0 = bp[0], b1 = bp[1];
        union { v16h h; uint32_t u[8]; } B;
        B.u[0] = b0.x; B.u[1] = b0.y; B.u[2] = b0.z; B.u[3] = b0.w;
        B.u[4] = b1.x; B.u[5] = b1.y; B.u[6] = b1.z; B.u[7] = b1.w;

        v8f acc = {};
        acc = __builtin_amdgcn_wmma_f32_16x16x32_f16(
            /*neg_a=*/false, A.h, /*neg_b=*/false, B.h,
            /*c_mod=*/(short)0, acc, /*reuse_a=*/false, /*reuse_b=*/false);

        // D layout: lane -> col (lane%16); VGPR r -> row r + 8*(lane/16)
#pragma unroll
        for (int r = 0; r < 8; ++r) {
            int mrow = r + 8 * half;
            z[((size_t)tile * 16 + mrow) * 1024 + (size_t)c * 16 + m] = (_Float16)acc[r];
        }
    }
}

// per-edge contraction + scatter-add: one wave per edge, lane = output channel
__global__ void __launch_bounds__(256) k_edge_msg(const int* __restrict__ ei,
                                                  const _Float16* __restrict__ h16,
                                                  const _Float16* __restrict__ z,
                                                  float* __restrict__ agg) {
    int e = blockIdx.x * 8 + (threadIdx.x >> 5);
    if (e >= N_EDGES) return;
    int lane = threadIdx.x & 31;
    int src = ei[e], dst = ei[N_EDGES + e];
    const _Float16* zr = z + (size_t)src * 1024;
    const _Float16* hr = h16 + (size_t)e * 32;
    float acc = 0.0f;
#pragma unroll
    for (int k = 0; k < 32; ++k)
        acc += (float)hr[k] * (float)zr[k * 32 + lane];
    atomicAdd(agg + (size_t)dst * 32 + lane, acc);
}

// x' = elu(agg/deg + x@wr + bc); also emit f16 copy for next layer's WMMA
__global__ void k_node_update(const float* __restrict__ agg, const float* __restrict__ deg,
                              const float* __restrict__ xin, const float* __restrict__ wr,
                              const float* __restrict__ bc, float* __restrict__ xout,
                              _Float16* __restrict__ x16o, int din) {
    int t = blockIdx.x * blockDim.x + threadIdx.x;
    if (t >= N_NODES * 32) return;
    int n = t >> 5, o = t & 31;
    float d = deg[n]; d = d < 1.0f ? 1.0f : d;
    float v = agg[t] / d + bc[o];
    for (int i = 0; i < din; ++i)
        v += xin[n * din + i] * wr[i * 32 + o];
    v = v > 0.0f ? v : (expf(v) - 1.0f);   // ELU
    xout[t] = v;
    x16o[t] = (_Float16)v;
}

// graph pooling (sum + count)
__global__ void k_pool(const float* __restrict__ xcur, const int* __restrict__ batch,
                       float* __restrict__ gsum, float* __restrict__ gcnt) {
    int t = blockIdx.x * blockDim.x + threadIdx.x;
    if (t >= N_NODES * 32) return;
    int n = t >> 5, o = t & 31;
    int g = batch[n];
    atomicAdd(&gsum[g * 32 + o], xcur[t]);
    if (o == 0) atomicAdd(&gcnt[g], 1.0f);
}

__global__ void k_final(const float* __restrict__ gsum, const float* __restrict__ gcnt,
                        const float* __restrict__ wfc, const float* __restrict__ bfc,
                        float* __restrict__ out) {
    int g = threadIdx.x;
    if (g >= N_GRAPHS) return;
    float c = gcnt[g]; c = c < 1.0f ? 1.0f : c;
    float s = 0.0f;
#pragma unroll
    for (int o = 0; o < 32; ++o) s += (gsum[g * 32 + o] / c) * wfc[o];
    out[g] = s + bfc[0];
}

// ---------------------------------------------------------------------------
extern "C" void kernel_launch(void* const* d_in, const int* in_sizes, int n_in,
                              void* d_out, int out_size, void* d_ws, size_t ws_size,
                              hipStream_t stream) {
    (void)in_sizes; (void)n_in; (void)out_size; (void)ws_size;
    const float* x     = (const float*)d_in[0];
    const float* ea    = (const float*)d_in[1];
    const int*   ei    = (const int*)d_in[2];
    const int*   batch = (const int*)d_in[3];
    const float* wfc   = (const float*)d_in[20];
    const float* bfc   = (const float*)d_in[21];

    uint8_t* ws = (uint8_t*)d_ws;
    size_t off = 0;
    auto alloc = [&](size_t bytes) -> void* {
        void* p = ws + off;
        off = (off + bytes + 255) & ~(size_t)255;
        return p;
    };
    _Float16* h16  = (_Float16*)alloc((size_t)N_EDGES * 32 * 2);
    _Float16* Z    = (_Float16*)alloc((size_t)N_NODES * 1024 * 2);
    _Float16* x16  = (_Float16*)alloc((size_t)N_NODES * 32 * 2);
    uint32_t* bpk  = (uint32_t*)alloc((size_t)64 * 32 * 8 * 4);
    float*    agg  = (float*)alloc((size_t)N_NODES * 32 * 4);
    float*    xa   = (float*)alloc((size_t)N_NODES * 32 * 4);
    float*    xb   = (float*)alloc((size_t)N_NODES * 32 * 4);
    float*    deg  = (float*)alloc((size_t)N_NODES * 4);
    float*    stats= (float*)alloc(64 * 4);
    float*    bnp  = (float*)alloc(64 * 4);
    float*    gsum = (float*)alloc((size_t)N_GRAPHS * 32 * 4);
    float*    gcnt = (float*)alloc((size_t)N_GRAPHS * 4);

    hipMemsetAsync(deg, 0, (size_t)N_NODES * 4, stream);
    k_deg<<<(N_EDGES + 255) / 256, 256, 0, stream>>>(ei, deg);
    k_cast_x16<<<(N_NODES * 32 + 255) / 256, 256, 0, stream>>>(x, x16, 16);

    const float* xin = x;
    float* xout = xa;
    for (int l = 0; l < 2; ++l) {
        const float* w1 = (const float*)d_in[4 + 8 * l];
        const float* b1 = (const float*)d_in[5 + 8 * l];
        const float* g  = (const float*)d_in[6 + 8 * l];
        const float* be = (const float*)d_in[7 + 8 * l];
        const float* w2 = (const float*)d_in[8 + 8 * l];
        const float* b2 = (const float*)d_in[9 + 8 * l];
        const float* wr = (const float*)d_in[10 + 8 * l];
        const float* bc = (const float*)d_in[11 + 8 * l];
        int din = (l == 0) ? 16 : 32;

        hipMemsetAsync(stats, 0, 64 * 4, stream);
        hipMemsetAsync(agg, 0, (size_t)N_NODES * 32 * 4, stream);

        k_edge_stats<<<160, 256, 0, stream>>>(ea, w1, b1, stats);
        k_bn_final<<<1, 32, 0, stream>>>(stats, g, be, bnp);
        k_edge_h16<<<(N_EDGES + 255) / 256, 256, 0, stream>>>(ea, w1, b1, bnp, h16);
        k_pack_b<<<(64 * 32 + 255) / 256, 256, 0, stream>>>(w2, b2, bpk, din);
        k_z_wmma<<<N_NODES / 16, 256, 0, stream>>>(x16, bpk, Z);
        k_edge_msg<<<N_EDGES / 8, 256, 0, stream>>>(ei, h16, Z, agg);
        k_node_update<<<(N_NODES * 32 + 255) / 256, 256, 0, stream>>>(
            agg, deg, xin, wr, bc, xout, x16, din);
        xin = xout;
        xout = xb;
    }

    hipMemsetAsync(gsum, 0, (size_t)N_GRAPHS * 32 * 4, stream);
    hipMemsetAsync(gcnt, 0, (size_t)N_GRAPHS * 4, stream);
    k_pool<<<(N_NODES * 32 + 255) / 256, 256, 0, stream>>>(xin, batch, gsum, gcnt);
    k_final<<<1, 128, 0, stream>>>(gsum, gcnt, wfc, bfc, (float*)d_out);
}
